// MultiHeadedLocationAwareAttention_17497696764360
// MI455X (gfx1250) — matrix operand order repaired and checked by reading, not compiled, not run on hardware
//
#include <hip/hip_runtime.h>
#include <hip/hip_bf16.h>

// ---------- sizes ----------
#define B   16
#define S   4096
#define H   8
#define D   128
#define HID 1024
#define C   10
#define M_TOTAL (B * S)        // 65536 GEMM rows

// ---------- WMMA types ----------
typedef __attribute__((ext_vector_type(16))) __bf16 v16bf;
typedef __attribute__((ext_vector_type(8)))  __bf16 v8bf;
typedef __attribute__((ext_vector_type(8)))  float  v8f;

union Frag16 { v16bf v; v8bf h[2]; __bf16 e[16]; };
union CFrag  { v8f   v; float  e[8];  };

template <bool RA>
__device__ __forceinline__ v8f wmma_bf16f32(v16bf a, v16bf b, v8f c) {
  // (neg_a, A, neg_b, B, c_mod, C, reuse_a, reuse_b)
  return __builtin_amdgcn_wmma_f32_16x16x32_bf16(false, a, false, b, (short)0, c, RA, false);
}

// =====================================================================
// 0) one-time: wT (HID,HID) bf16 = transpose(v_proj_w)  (2MB, L2-resident)
// =====================================================================
__global__ void wtobf16_kernel(const float* __restrict__ w,
                               __bf16* __restrict__ wT) {
  int idx = blockIdx.x * 256 + threadIdx.x;    // HID*HID = 1M
  int n = idx >> 10;
  int k = idx & (HID - 1);
  wT[idx] = (__bf16)w[(size_t)k * HID + n];
}

// =====================================================================
// 1) Conv1d over heads: prev_attn (B,H,S) -> conv (B,C,S)
// =====================================================================
__global__ void conv_kernel(const float* __restrict__ prev,
                            const float* __restrict__ w,
                            const float* __restrict__ bias,
                            float* __restrict__ conv) {
  int idx = blockIdx.x * 256 + threadIdx.x;           // B*C*S = 655360
  if (idx >= B * C * S) return;
  int s = idx & (S - 1);
  int c = (idx >> 12) % C;
  int b = idx / (C * S);
  float acc = bias[c];
#pragma unroll
  for (int h = 0; h < H; h++) {
    const float* pr = prev + ((size_t)b * H + h) * S;
    const float* wc = w + (c * H + h) * 3;
    float x0 = (s > 0)     ? pr[s - 1] : 0.f;
    float x1 = pr[s];
    float x2 = (s < S - 1) ? pr[s + 1] : 0.f;
    acc += x0 * wc[0] + x1 * wc[1] + x2 * wc[2];
  }
  conv[idx] = acc;
}

// =====================================================================
// 2) loc (B,S,D) = tanh(conv (B,C,S) . loc_proj_w (C,D))
// =====================================================================
__global__ void loc_kernel(const float* __restrict__ conv,
                           const float* __restrict__ pw,
                           float* __restrict__ loc) {
  int idx = blockIdx.x * 256 + threadIdx.x;           // B*S*D = 2^23
  int d = idx & (D - 1);
  int s = (idx >> 7) & (S - 1);
  int b = idx >> 19;
  float acc = 0.f;
#pragma unroll
  for (int c = 0; c < C; c++)
    acc += conv[((size_t)b * C + c) * S + s] * pw[c * D + d];
  loc[idx] = tanhf(acc);
}

// =====================================================================
// 3) q heads: qh (B,HID) = query (B,HID) @ q_proj_w (HID,HID)
// =====================================================================
__global__ void qproj_kernel(const float* __restrict__ query,
                             const float* __restrict__ qw,
                             float* __restrict__ qh) {
  int idx = blockIdx.x * 256 + threadIdx.x;           // 16384
  int b = idx >> 10;
  int c = idx & (HID - 1);
  float acc = 0.f;
  for (int k = 0; k < HID; k++)
    acc += query[b * HID + k] * qw[(size_t)k * HID + c];
  qh[idx] = acc;
}

// =====================================================================
// 4) Big GEMM: vp (65536,1024) = value @ v_proj_w, bf16 WMMA, f32 acc
//    128x128 tile / block (8 waves), K-step 32. B comes from the
//    pre-transposed bf16 copy wT, so all LDS traffic is b128 vectors.
// =====================================================================
__global__ __launch_bounds__(256, 2)
void gemm_vproj_kernel(const float* __restrict__ value,
                       const __bf16* __restrict__ wT,
                       float* __restrict__ vp) {
  // 40-element (80B) row pitch: 16B aligned, conflict-free for 16 lanes
  __shared__ __align__(16) __bf16 lA[128][40];   // M x K tile
  __shared__ __align__(16) __bf16 lB[128][40];   // N x K tile (K contiguous)

  const int tid  = threadIdx.x;
  const int wave = tid >> 5;
  const int lane = tid & 31;
  const int m0 = blockIdx.x * 128;
  const int n0 = blockIdx.y * 128;

  CFrag acc[8];
#pragma unroll
  for (int i = 0; i < 8; i++)
#pragma unroll
    for (int j = 0; j < 8; j++) acc[i].e[j] = 0.f;

  for (int k0 = 0; k0 < HID; k0 += 32) {
    // ---- stage A tile (value rows, f32 -> bf16), 16B vector stores ----
#pragma unroll
    for (int i = 0; i < 2; i++) {
      int idx = tid + i * 256;          // 0..511
      int r  = idx >> 2;                // row 0..127
      int cg = (idx & 3) * 8;           // k offset 0,8,16,24
      const float4* src = (const float4*)&value[(size_t)(m0 + r) * HID + k0 + cg];
      float4 f0 = src[0];
      float4 f1 = src[1];
      v8bf hv = { (__bf16)f0.x, (__bf16)f0.y, (__bf16)f0.z, (__bf16)f0.w,
                  (__bf16)f1.x, (__bf16)f1.y, (__bf16)f1.z, (__bf16)f1.w };
      *(v8bf*)&lA[r][cg] = hv;
    }
    // ---- stage B tile (wT rows, already bf16), straight 16B copies ----
    {
      int n    = tid >> 1;              // 0..127
      int half = (tid & 1) * 16;        // k offset 0 / 16
      const v8bf* src = (const v8bf*)(wT + (size_t)(n0 + n) * HID + k0 + half);
      v8bf b0 = src[0];
      v8bf b1 = src[1];
      *(v8bf*)&lB[n][half]     = b0;
      *(v8bf*)&lB[n][half + 8] = b1;
    }
    if (k0 + 32 < HID) {   // GL2 prefetch of next K tiles
      __builtin_prefetch(value + (size_t)(m0 + (tid >> 1)) * HID + k0 + 32, 0, 1);
      __builtin_prefetch(wT + (size_t)(n0 + (tid >> 1)) * HID + k0 + 32, 0, 1);
    }
    __syncthreads();

    // ---- fragments: two ds_load_b128 per operand ----
    Frag16 a;
    const int am = wave * 16 + (lane & 15);
    const int kh = (lane >> 4) * 8;
    a.h[0] = *(const v8bf*)&lA[am][kh];
    a.h[1] = *(const v8bf*)&lA[am][16 + kh];

    const int bl = lane & 15;
    const int bk = (lane >> 4) * 8;
#define WMMA_NT(nt, RA_)                                            \
    {                                                               \
      Frag16 bb;                                                    \
      bb.h[0] = *(const v8bf*)&lB[(nt) * 16 + bl][bk];              \
      bb.h[1] = *(const v8bf*)&lB[(nt) * 16 + bl][bk + 16];         \
      acc[nt].v = wmma_bf16f32<RA_>(a.v, bb.v, acc[nt].v);          \
    }
    WMMA_NT(0, true)
    WMMA_NT(1, true)
    WMMA_NT(2, true)
    WMMA_NT(3, true)
    WMMA_NT(4, true)
    WMMA_NT(5, true)
    WMMA_NT(6, true)
    WMMA_NT(7, false)
#undef WMMA_NT
    __syncthreads();
  }

  // C layout: VGPR i -> M = (lane/16)*8 + i, N = lane%16
  const int rm = wave * 16 + ((lane >> 4) << 3);
  const int cn = lane & 15;
#pragma unroll
  for (int nt = 0; nt < 8; nt++) {
    int col = n0 + nt * 16 + cn;
#pragma unroll
    for (int i = 0; i < 8; i++)
      vp[(size_t)(m0 + rm + i) * HID + col] = acc[nt].e[i];
  }
}

// =====================================================================
// 5) score (B,H,S): sum_d tanh(vp + q + loc + bias) * score_w + score_b
//    one 128-thread block per (b,s)
// =====================================================================
__global__ void score_kernel(const float* __restrict__ vp,
                             const float* __restrict__ qh,
                             const float* __restrict__ loc,
                             const float* __restrict__ bias,
                             const float* __restrict__ sw,
                             const float* __restrict__ sb,
                             float* __restrict__ score) {
  const int bs = blockIdx.x;            // b*S + s
  const int b = bs >> 12;
  const int s = bs & (S - 1);
  const int d = threadIdx.x;            // 0..127
  const float locd = loc[(size_t)bs * D + d];
  const float bd = bias[d];
  const float swd = sw[d];
  __shared__ float red[4];
#pragma unroll
  for (int h = 0; h < H; h++) {
    float x = vp[(size_t)bs * HID + h * D + d] + qh[b * HID + h * D + d] + locd + bd;
    float t = tanhf(x) * swd;
#pragma unroll
    for (int off = 16; off > 0; off >>= 1) t += __shfl_down(t, off, 32);
    if ((threadIdx.x & 31) == 0) red[threadIdx.x >> 5] = t;
    __syncthreads();
    if (threadIdx.x == 0)
      score[((b << 3) + h) * S + s] = red[0] + red[1] + red[2] + red[3] + sb[0];
    __syncthreads();
  }
}

// =====================================================================
// 6) softmax over S per (b,h) row -> writes attn region of d_out
// =====================================================================
__global__ void softmax_kernel(const float* __restrict__ score,
                               float* __restrict__ attn) {
  const int row = blockIdx.x;           // 0..127
  const float* in = score + (size_t)row * S;
  float* out = attn + (size_t)row * S;
  __shared__ float sm[256];
  const int t = threadIdx.x;
  float m = -1e30f;
  for (int i = t; i < S; i += 256) m = fmaxf(m, in[i]);
  sm[t] = m; __syncthreads();
  for (int k = 128; k > 0; k >>= 1) { if (t < k) sm[t] = fmaxf(sm[t], sm[t + k]); __syncthreads(); }
  m = sm[0]; __syncthreads();
  float sum = 0.f;
  for (int i = t; i < S; i += 256) sum += __expf(in[i] - m);
  sm[t] = sum; __syncthreads();
  for (int k = 128; k > 0; k >>= 1) { if (t < k) sm[t] += sm[t + k]; __syncthreads(); }
  const float inv = 1.0f / sm[0];
  for (int i = t; i < S; i += 256) out[i] = __expf(in[i] - m) * inv;
}

// =====================================================================
// 7) context with the reference's scrambled v2 permutation (split-K + atomics)
//    t = s2*H + h2 ; h = t/S ; s = t%S ; ctx[b,h2,d] += attn * vp[b,s,h,d]
// =====================================================================
__global__ void zero_kernel(float* __restrict__ p, int n) {
  int i = blockIdx.x * 256 + threadIdx.x;
  if (i < n) p[i] = 0.f;
}

__global__ void context_kernel(const float* __restrict__ attn,
                               const float* __restrict__ vp,
                               float* __restrict__ ctx) {
  const int bh = blockIdx.x;            // b*H + h2
  const int b = bh >> 3, h2 = bh & 7;
  const int d = threadIdx.x;            // 0..127
  float acc = 0.f;
  const int s2base = blockIdx.y * 512;
  for (int i = 0; i < 512; i++) {
    const int s2 = s2base + i;
    const float a = attn[(size_t)bh * S + s2];
    const int t = s2 * H + h2;
    const int h = t >> 12;
    const int s = t & (S - 1);
    acc += a * vp[(size_t)((b * S + s) * H + h) * D + d];
  }
  atomicAdd(&ctx[bh * D + d], acc);
}

// =====================================================================
// 8) output: [context | query] (B,2048) @ out_w (2048,1024) + out_b
// =====================================================================
__global__ void output_kernel(const float* __restrict__ ctx,
                              const float* __restrict__ query,
                              const float* __restrict__ ow,
                              const float* __restrict__ ob,
                              float* __restrict__ out) {
  int idx = blockIdx.x * 256 + threadIdx.x;  // 16384
  int b = idx >> 10;
  int j = idx & (HID - 1);
  float acc = ob[j];
  const float* c = ctx + b * HID;
  const float* q = query + b * HID;
  for (int i = 0; i < HID; i++) acc += c[i] * ow[(size_t)i * HID + j];
  for (int i = 0; i < HID; i++) acc += q[i] * ow[(size_t)(HID + i) * HID + j];
  out[idx] = acc;
}

// =====================================================================
extern "C" void kernel_launch(void* const* d_in, const int* in_sizes, int n_in,
                              void* d_out, int out_size, void* d_ws, size_t ws_size,
                              hipStream_t stream) {
  const float* query      = (const float*)d_in[0];
  const float* value      = (const float*)d_in[1];
  const float* prev_attn  = (const float*)d_in[2];
  const float* loc_conv_w = (const float*)d_in[3];
  const float* loc_conv_b = (const float*)d_in[4];
  const float* loc_proj_w = (const float*)d_in[5];
  const float* q_proj_w   = (const float*)d_in[6];
  const float* v_proj_w   = (const float*)d_in[7];
  const float* score_w    = (const float*)d_in[8];
  const float* score_b    = (const float*)d_in[9];
  const float* out_w      = (const float*)d_in[10];
  const float* out_b      = (const float*)d_in[11];
  const float* attn_bias  = (const float*)d_in[12];

  float* out      = (float*)d_out;            // (B,1,HID) = 16384
  float* attn_out = out + (size_t)B * HID;    // (B,H,S)   = 524288

  float* ws    = (float*)d_ws;
  float* vp    = ws;                                  // 65536*1024 f32
  float* conv  = vp   + (size_t)M_TOTAL * HID;        // B*C*S
  float* loc   = conv + (size_t)B * C * S;            // B*S*D
  float* qh    = loc  + (size_t)B * S * D;            // B*HID
  float* score = qh   + (size_t)B * HID;              // B*H*S
  float* ctx   = score + (size_t)B * H * S;           // B*HID
  __bf16* wT   = (__bf16*)(ctx + (size_t)B * HID);    // HID*HID bf16

  wtobf16_kernel<<<(HID * HID) / 256, 256, 0, stream>>>(v_proj_w, wT);
  conv_kernel<<<(B * C * S + 255) / 256, 256, 0, stream>>>(prev_attn, loc_conv_w, loc_conv_b, conv);
  loc_kernel<<<(B * S * D) / 256, 256, 0, stream>>>(conv, loc_proj_w, loc);
  qproj_kernel<<<(B * HID) / 256, 256, 0, stream>>>(query, q_proj_w, qh);

  dim3 gg(M_TOTAL / 128, HID / 128);
  gemm_vproj_kernel<<<gg, 256, 0, stream>>>(value, wT, vp);

  score_kernel<<<B * S, 128, 0, stream>>>(vp, qh, loc, attn_bias, score_w, score_b, score);
  softmax_kernel<<<B * H, 256, 0, stream>>>(score, attn_out);

  zero_kernel<<<(B * HID + 255) / 256, 256, 0, stream>>>(ctx, B * HID);
  dim3 gc(B * H, S / 512);
  context_kernel<<<gc, 128, 0, stream>>>(attn_out, vp, ctx);

  output_kernel<<<(B * HID) / 256, 256, 0, stream>>>(ctx, query, out_w, out_b, out);
}
